// GATEncoder_32195074851467
// MI455X (gfx1250) — matrix operand order repaired and checked by reading, not compiled, not run on hardware
//
#include <hip/hip_runtime.h>
#include <hip/hip_bf16.h>
#include <stdint.h>

typedef _Float16 f16;
typedef __attribute__((ext_vector_type(16))) _Float16 v16h;
typedef __attribute__((ext_vector_type(8)))  float    v8f;

#define N_ATOMS  200000
#define N_EDGES  800000
#define ATOM_FDIM 133
#define HIDDEN   256
#define NUM_MOLS 8000
#define ALPHA    0.2f

// ---------------- conversion kernels (f32 -> f16, padded K) ----------------

// wT[n][k] = f16(W[k][n]), zero-padded for k >= K.  (B operand, pre-transposed
// so the WMMA B fragment is two contiguous 16B loads per lane.)
__global__ void conv_w_f16(const float* __restrict__ W, f16* __restrict__ wT,
                           int K, int KP) {
  int i = blockIdx.x * 256 + threadIdx.x;      // HIDDEN*KP threads, exact grid
  int k = i % KP;
  int n = i / KP;
  wT[(size_t)n * KP + k] = (k < K) ? (f16)W[(size_t)k * HIDDEN + n] : (f16)0.f;
}

// xh[row][k] = f16(src[row][k]), zero-padded for k >= Ksrc.
__global__ void conv_x_f16(const float* __restrict__ src, f16* __restrict__ dst,
                           int Ksrc, int KP) {
  size_t i = (size_t)blockIdx.x * 256 + threadIdx.x;   // N_ATOMS*KP threads
  int    k   = (int)(i % KP);
  size_t row = i / KP;
  dst[i] = (k < Ksrc) ? (f16)src[row * Ksrc + k] : (f16)0.f;
}

// ---------------- WMMA GEMM: h[N][256] = xh[N][KP] * wT^T ----------------
// One wave per 16x16 output tile. f16 inputs, f32 accumulation.
__global__ void gat_gemm_wmma(const f16* __restrict__ xh,
                              const f16* __restrict__ wT,
                              float* __restrict__ h, int KP) {
  const int wave = blockIdx.x * 8 + (threadIdx.x >> 5);
  const int lane = threadIdx.x & 31;
  const int mt   = wave >> 4;        // 0..12499  (M tile)
  const int nt   = wave & 15;        // 0..15     (N tile)
  const int r15  = lane & 15;
  const int hh   = lane >> 4;        // half-wave select

  // A fragment (16-bit A 16x32 layout): lanes 0-15 -> K+0..7 / K+16..23,
  // lanes 16-31 -> K+8..15 / K+24..31, row = mt*16 + (lane&15).
  const f16* aptr = xh + (size_t)(mt * 16 + r15) * KP + hh * 8;
  // B fragment: col = nt*16 + (lane&15); lanes 0-15 hold K+0..7 / K+16..23,
  // lanes 16-31 hold K+8..15 / K+24..31 (same k order as A).
  const f16* bptr = wT + (size_t)(nt * 16 + r15) * KP + hh * 8;

  v8f acc = {};
  for (int kk = 0; kk < KP; kk += 32) {
    union { uint4 q[2]; v16h v; } A, B;
    A.q[0] = *reinterpret_cast<const uint4*>(aptr);
    A.q[1] = *reinterpret_cast<const uint4*>(aptr + 16);
    B.q[0] = *reinterpret_cast<const uint4*>(bptr);
    B.q[1] = *reinterpret_cast<const uint4*>(bptr + 16);
    __builtin_prefetch(aptr + 32, 0, 0);   // global_prefetch_b8, next K slab
    acc = __builtin_amdgcn_wmma_f32_16x16x32_f16(
        /*neg_a=*/false, A.v, /*neg_b=*/false, B.v,
        /*c_mod=*/(short)0, acc, /*reuse_a=*/false, /*reuse_b=*/false);
    aptr += 32;
    bptr += 32;
  }

  // C/D layout: VGPR r -> (M = r + 8*hh, N = lane&15)
  float* hrow = h + (size_t)(mt * 16 + hh * 8) * HIDDEN + nt * 16 + r15;
#pragma unroll
  for (int r = 0; r < 8; ++r)
    hrow[(size_t)r * HIDDEN] = acc[r];
}

// ---------------- per-row attention dots: hs = h.a_s, ht = h.a_t ----------
__global__ void gat_rowdots(const float* __restrict__ h,
                            const float* __restrict__ a,
                            float* __restrict__ hs, float* __restrict__ ht) {
  int row  = blockIdx.x * 8 + (threadIdx.x >> 5);
  int lane = threadIdx.x & 31;
  const float* hr = h + (size_t)row * HIDDEN;
  float s = 0.f, t = 0.f;
#pragma unroll
  for (int j = 0; j < 8; ++j) {
    int c = lane + 32 * j;
    float v = hr[c];
    s += v * a[c];
    t += v * a[HIDDEN + c];
  }
#pragma unroll
  for (int m = 16; m; m >>= 1) {
    s += __shfl_xor(s, m, 32);
    t += __shfl_xor(t, m, 32);
  }
  if (lane == 0) { hs[row] = s; ht[row] = t; }
}

// ---------------- edge pass 1: att = exp(leaky_relu), denom scatter -------
__global__ void gat_edge_att(const float* __restrict__ hs,
                             const float* __restrict__ ht,
                             const int* __restrict__ src,
                             const int* __restrict__ tgt,
                             float* __restrict__ att, float* __restrict__ den) {
  int e = blockIdx.x * blockDim.x + threadIdx.x;
  if (e >= N_EDGES) return;
  int t = tgt[e];
  float v = hs[src[e]] + ht[t];
  v = (v > 0.f) ? v : ALPHA * v;
  float a = __expf(v);
  att[e] = a;
  atomicAdd(&den[t], a);
}

// ---------------- edge pass 2: hp[tgt] += h[src] * att_n (wave/edge) ------
__global__ void gat_edge_aggr(const float* __restrict__ h,
                              const float* __restrict__ att,
                              const float* __restrict__ den,
                              const int* __restrict__ src,
                              const int* __restrict__ tgt,
                              float* __restrict__ hp) {
  int e = blockIdx.x * 8 + (threadIdx.x >> 5);
  int lane = threadIdx.x & 31;
  int s = src[e], t = tgt[e];
  float w = att[e] / (den[t] + 1e-8f);
  const float* hsrc = h + (size_t)s * HIDDEN;   // coalesced 128B gathers
  float* hpd = hp + (size_t)t * HIDDEN;
#pragma unroll
  for (int j = 0; j < 8; ++j) {
    int c = lane + 32 * j;
    atomicAdd(&hpd[c], hsrc[c] * w);            // global_atomic_add_f32
  }
}

// ---------------- residual + ELU, written back into x_cur -----------------
__global__ void gat_finalize(const float* __restrict__ hp,
                             float* __restrict__ xc, int residual) {
  size_t i = (size_t)blockIdx.x * 256 + threadIdx.x;
  float v = hp[i] + (residual ? xc[i] : 0.f);
  xc[i] = (v > 0.f) ? v : (__expf(v) - 1.f);
}

// ---------------- molecule pooling ----------------------------------------
__global__ void gat_pool_add(const float* __restrict__ x,
                             const int* __restrict__ mol,
                             float* __restrict__ out, float* __restrict__ cnt) {
  int atom = blockIdx.x * 8 + (threadIdx.x >> 5);
  int lane = threadIdx.x & 31;
  int m = mol[atom];
  const float* xr = x + (size_t)atom * HIDDEN;
  float* orow = out + (size_t)m * HIDDEN;
#pragma unroll
  for (int j = 0; j < 8; ++j) {
    int c = lane + 32 * j;
    atomicAdd(&orow[c], xr[c]);
  }
  if (lane == 0) atomicAdd(&cnt[m], 1.f);
}

__global__ void gat_pool_div(float* __restrict__ out,
                             const float* __restrict__ cnt) {
  int i = blockIdx.x * 256 + threadIdx.x;       // NUM_MOLS*HIDDEN, exact
  float c = cnt[i >> 8];
  out[i] = (c > 0.f) ? out[i] / c : 0.f;
}

// ---------------- host orchestration --------------------------------------
extern "C" void kernel_launch(void* const* d_in, const int* in_sizes, int n_in,
                              void* d_out, int out_size, void* d_ws,
                              size_t ws_size, hipStream_t stream) {
  (void)in_sizes; (void)n_in;
  const float* f_atoms = (const float*)d_in[0];
  const int*   ei      = (const int*)d_in[1];
  const int*   mol     = (const int*)d_in[2];
  const float* Ws[3]   = {(const float*)d_in[3], (const float*)d_in[5],
                          (const float*)d_in[7]};
  const float* as[3]   = {(const float*)d_in[4], (const float*)d_in[6],
                          (const float*)d_in[8]};
  const int* src = ei;
  const int* tgt = ei + N_EDGES;

  // workspace layout (all 256B aligned)
  char*  ws   = (char*)d_ws;
  size_t szH  = (size_t)N_ATOMS * HIDDEN * sizeof(float);   // 204.8 MB
  size_t szXH = (size_t)N_ATOMS * HIDDEN * sizeof(f16);     // 102.4 MB
  float* h   = (float*)(ws);
  float* hp  = (float*)(ws + szH);
  float* xc  = (float*)(ws + 2 * szH);
  f16*   xh  = (f16*)  (ws + 3 * szH);
  f16*   wT  = (f16*)  (ws + 3 * szH + szXH);
  float* hs  = (float*)(ws + 3 * szH + szXH + 131072);
  float* ht  = hs + N_ATOMS;
  float* den = ht + N_ATOMS;
  float* att = den + N_ATOMS;
  float* cnt = att + N_EDGES;
  size_t need = 3 * szH + szXH + 131072 +
                sizeof(float) * ((size_t)3 * N_ATOMS + N_EDGES + NUM_MOLS);
  if (ws_size < need) return;

  for (int L = 0; L < 3; ++L) {
    const int   K   = (L == 0) ? ATOM_FDIM : HIDDEN;
    const int   KP  = (L == 0) ? 160 : HIDDEN;           // pad K to mult. of 32
    const float* xin = (L == 0) ? f_atoms : xc;

    conv_w_f16<<<(HIDDEN * KP) / 256, 256, 0, stream>>>(Ws[L], wT, K, KP);
    conv_x_f16<<<(int)(((size_t)N_ATOMS * KP) / 256), 256, 0, stream>>>(
        xin, xh, K, KP);
    gat_gemm_wmma<<<(N_ATOMS / 16) * 16 / 8, 256, 0, stream>>>(xh, wT, h, KP);
    gat_rowdots<<<N_ATOMS / 8, 256, 0, stream>>>(h, as[L], hs, ht);

    hipMemsetAsync(den, 0, (size_t)N_ATOMS * sizeof(float), stream);
    hipMemsetAsync(hp, 0, szH, stream);
    gat_edge_att<<<(N_EDGES + 255) / 256, 256, 0, stream>>>(hs, ht, src, tgt,
                                                            att, den);
    gat_edge_aggr<<<N_EDGES / 8, 256, 0, stream>>>(h, att, den, src, tgt, hp);
    gat_finalize<<<(int)((size_t)N_ATOMS * HIDDEN / 256), 256, 0, stream>>>(
        hp, xc, L > 0);
  }

  hipMemsetAsync(d_out, 0, (size_t)out_size * sizeof(float), stream);
  hipMemsetAsync(cnt, 0, (size_t)NUM_MOLS * sizeof(float), stream);
  gat_pool_add<<<N_ATOMS / 8, 256, 0, stream>>>(xc, mol, (float*)d_out, cnt);
  gat_pool_div<<<(NUM_MOLS * HIDDEN) / 256, 256, 0, stream>>>((float*)d_out,
                                                              cnt);
}
